// StockGRU_45887430591112
// MI455X (gfx1250) — matrix operand order repaired and checked by reading, not compiled, not run on hardware
//
#include <hip/hip_runtime.h>

#define HID 128
#define BATCH 64
#define TSTEPS 4096
#define NTHREADS 1024

typedef __bf16 v16bf __attribute__((ext_vector_type(16)));
typedef __bf16 v8bf  __attribute__((ext_vector_type(8)));
typedef float  v8f   __attribute__((ext_vector_type(8)));

union AFrag { v16bf v; struct { v8bf lo, hi; } h; };

__device__ __forceinline__ v8f zero8() {
  v8f z = {0.f, 0.f, 0.f, 0.f, 0.f, 0.f, 0.f, 0.f};
  return z;
}

__device__ __forceinline__ float sigm(float x) { return 1.0f / (1.0f + __expf(-x)); }
__device__ __forceinline__ float tanh_f(float x) { return 1.0f - 2.0f / (__expf(2.0f * x) + 1.0f); }

// A fragment: 16x32 bf16 tile of X (row-major, row stride HID).
// Contiguous k order used here: lane holds row M = lane&15; elements 0..15 hold
// K = kbase + half*16 .. +16 (matches B fragment).
__device__ __forceinline__ v16bf load_a(const __bf16* xrow, int kbase, int half) {
  AFrag f;
  f.h.lo = *(const v8bf*)(xrow + kbase + half * 16);
  f.h.hi = *(const v8bf*)(xrow + kbase + half * 16 + 8);
  return f.v;
}

// B fragment: 32x16 bf16 tile of W^T, W row-major [384][128] (K contiguous).
// Lane holds tile col N = lane&15 -> W row `ncol`; elements 0..15 hold
// K = kbase + half*16 .. +16 (one contiguous 32B run).
__device__ __forceinline__ v16bf load_b(const __bf16* w, int ncol, int kbase, int half) {
  return *(const v16bf*)(w + ncol * HID + kbase + half * 16);
}

#define WMMA_BF16(a, b, c) \
  __builtin_amdgcn_wmma_f32_16x16x32_bf16(false, (a), false, (b), (short)0, (c), false, false)

extern "C" __global__ __launch_bounds__(NTHREADS, 1)
void gru2_fused_kernel(const float* __restrict__ data,    // [B, T, 1] -> [B][T]
                       const float* __restrict__ hidden,  // [2, B, H]
                       const float* __restrict__ w_ih0,   // [3H, 1]
                       const float* __restrict__ w_hh0,   // [3H, H]
                       const float* __restrict__ b_ih0,   // [3H]
                       const float* __restrict__ b_hh0,   // [3H]
                       const float* __restrict__ w_ih1,   // [3H, H]
                       const float* __restrict__ w_hh1,   // [3H, H]
                       const float* __restrict__ b_ih1,   // [3H]
                       const float* __restrict__ b_hh1,   // [3H]
                       const float* __restrict__ fc_w,    // [1, H]
                       const float* __restrict__ fc_b,    // [1]
                       float* __restrict__ out)           // [B*T] ++ [2*B*H]
{
  extern __shared__ __bf16 smem[];
  __bf16* w0  = smem;                       // w_hh0 bf16 [384*128]   96 KB
  __bf16* w1  = smem + 384 * 128;           // w_hh1 bf16             96 KB
  __bf16* wi1 = smem + 2 * 384 * 128;       // w_ih1 bf16             96 KB
  __bf16* h0s = smem + 3 * 384 * 128;       // h0 bf16 [64*128]       16 KB
  __bf16* h1s = h0s + 64 * 128;             // h1 bf16 [64*128]       16 KB

  const int tid = threadIdx.x;

  // One-time: convert resident weights f32 -> bf16 into LDS.
  for (int i = tid; i < 384 * 128; i += NTHREADS) {
    w0[i]  = (__bf16)w_hh0[i];
    w1[i]  = (__bf16)w_hh1[i];
    wi1[i] = (__bf16)w_ih1[i];
  }
  // Zero the out_data region (atomic accumulation target). Deterministic per call.
  for (int i = tid; i < BATCH * TSTEPS; i += NTHREADS) out[i] = 0.0f;

  const int wave = tid >> 5;        // 0..31
  const int lane = tid & 31;
  const int lh   = lane & 15;
  const int half = lane >> 4;
  const int m    = wave >> 3;       // batch tile 0..3 (rows m*16..m*16+15)
  const int jg   = wave & 7;        // hidden column group 0..7
  const int j    = jg * 16 + lh;    // hidden column 0..127 owned by this lane
  const int rowbase = m * 16 + half * 8;   // + v (0..7) -> batch row (C/D layout)

  // Per-lane gate constants for column j.
  const float wih0_r = w_ih0[j], wih0_z = w_ih0[HID + j], wih0_n = w_ih0[2 * HID + j];
  const float bih0_r = b_ih0[j], bih0_z = b_ih0[HID + j], bih0_n = b_ih0[2 * HID + j];
  const float bhh0_r = b_hh0[j], bhh0_z = b_hh0[HID + j], bhh0_n = b_hh0[2 * HID + j];
  const float bih1_r = b_ih1[j], bih1_z = b_ih1[HID + j], bih1_n = b_ih1[2 * HID + j];
  const float bhh1_r = b_hh1[j], bhh1_z = b_hh1[HID + j], bhh1_n = b_hh1[2 * HID + j];
  const float fcw = fc_w[j];
  const float fcb = fc_b[0];

  // f32 master copies of this wave's h patches (rows rowbase+v, col j).
  float h0m[8], h1m[8];
#pragma unroll
  for (int v = 0; v < 8; ++v) {
    const int b = rowbase + v;
    h0m[v] = hidden[b * HID + j];
    h1m[v] = hidden[BATCH * HID + b * HID + j];
    h0s[b * HID + j] = (__bf16)h0m[v];
    h1s[b * HID + j] = (__bf16)h1m[v];
  }
  __syncthreads();

  const __bf16* arow0 = h0s + (m * 16 + lh) * HID;  // A rows: M = lane&15 in tile
  const __bf16* arow1 = h1s + (m * 16 + lh) * HID;

  for (int t = 0; t < TSTEPS; ++t) {
    // Prefetch this step's scalar inputs x[b] (layer-0 input dim is 1).
    float xv[8];
#pragma unroll
    for (int v = 0; v < 8; ++v) xv[v] = data[(rowbase + v) * TSTEPS + t];

    // ---------------- layer 0: hg = h0 @ w_hh0^T ----------------
    v8f ar = zero8(), az = zero8(), an = zero8();
#pragma unroll
    for (int kk = 0; kk < 4; ++kk) {
      const int kb = kk * 32;
      const v16bf a = load_a(arow0, kb, half);
      ar = WMMA_BF16(a, load_b(w0, j, kb, half), ar);
      az = WMMA_BF16(a, load_b(w0, 128 + j, kb, half), az);
      an = WMMA_BF16(a, load_b(w0, 256 + j, kb, half), an);
    }
#pragma unroll
    for (int v = 0; v < 8; ++v) {
      const float xr = xv[v] * wih0_r + bih0_r;
      const float xz = xv[v] * wih0_z + bih0_z;
      const float xn = xv[v] * wih0_n + bih0_n;
      const float r = sigm(xr + ar[v] + bhh0_r);
      const float z = sigm(xz + az[v] + bhh0_z);
      const float n = tanh_f(xn + r * (an[v] + bhh0_n));
      h0m[v] = (1.0f - z) * n + z * h0m[v];
    }
    __syncthreads();                       // all waves done reading old h0s
#pragma unroll
    for (int v = 0; v < 8; ++v)
      h0s[(rowbase + v) * HID + j] = (__bf16)h0m[v];   // publish y0_t
    __syncthreads();                       // y0_t visible to all waves

    // ------- layer 1: K=256 over [h1 | y0] x [w_hh1 ; w_ih1] -------
    v8f ar1 = zero8(), az1 = zero8(), axn = zero8(), ahn = zero8();
#pragma unroll
    for (int kk = 0; kk < 4; ++kk) {       // recurrent part (h1, w_hh1)
      const int kb = kk * 32;
      const v16bf a = load_a(arow1, kb, half);
      ar1 = WMMA_BF16(a, load_b(w1, j, kb, half), ar1);
      az1 = WMMA_BF16(a, load_b(w1, 128 + j, kb, half), az1);
      ahn = WMMA_BF16(a, load_b(w1, 256 + j, kb, half), ahn);
    }
#pragma unroll
    for (int kk = 0; kk < 4; ++kk) {       // input part (y0, w_ih1)
      const int kb = kk * 32;
      const v16bf a = load_a(arow0, kb, half);
      ar1 = WMMA_BF16(a, load_b(wi1, j, kb, half), ar1);
      az1 = WMMA_BF16(a, load_b(wi1, 128 + j, kb, half), az1);
      axn = WMMA_BF16(a, load_b(wi1, 256 + j, kb, half), axn);
    }

    float part[8];
#pragma unroll
    for (int v = 0; v < 8; ++v) {
      const float r = sigm(ar1[v] + bih1_r + bhh1_r);
      const float z = sigm(az1[v] + bih1_z + bhh1_z);
      const float n = tanh_f(axn[v] + bih1_n + r * (ahn[v] + bhh1_n));
      const float hn = (1.0f - z) * n + z * h1m[v];
      h1m[v] = hn;
      const float y = hn > 0.0f ? hn : 0.0f;   // ReLU before FC
      part[v] = y * fcw;
    }
    __syncthreads();                       // all waves done reading old h1s
#pragma unroll
    for (int v = 0; v < 8; ++v)
      h1s[(rowbase + v) * HID + j] = (__bf16)h1m[v];

    // FC: reduce this wave's 16 columns, then one atomic per row.
#pragma unroll
    for (int v = 0; v < 8; ++v) {
      float s = part[v];
      for (int mk = 1; mk < 16; mk <<= 1) s += __shfl_xor(s, mk, 32);
      part[v] = s;
    }
    if (lh == 0) {
      const float add = (jg == 0) ? fcb : 0.0f;
#pragma unroll
      for (int v = 0; v < 8; ++v)
        atomicAdd(out + (rowbase + v) * TSTEPS + t, part[v] + add);
    }
    __syncthreads();                       // h1_t visible for next step
  }

  // Final hidden states: out[B*T ..] = [2, B, H].
#pragma unroll
  for (int v = 0; v < 8; ++v) {
    const int b = rowbase + v;
    out[BATCH * TSTEPS + b * HID + j] = h0m[v];
    out[BATCH * TSTEPS + BATCH * HID + b * HID + j] = h1m[v];
  }
}

extern "C" void kernel_launch(void* const* d_in, const int* in_sizes, int n_in,
                              void* d_out, int out_size, void* d_ws, size_t ws_size,
                              hipStream_t stream) {
  (void)in_sizes; (void)n_in; (void)out_size; (void)d_ws; (void)ws_size;
  const float* data   = (const float*)d_in[0];
  const float* hidden = (const float*)d_in[1];
  const float* w_ih0  = (const float*)d_in[2];
  const float* w_hh0  = (const float*)d_in[3];
  const float* b_ih0  = (const float*)d_in[4];
  const float* b_hh0  = (const float*)d_in[5];
  const float* w_ih1  = (const float*)d_in[6];
  const float* w_hh1  = (const float*)d_in[7];
  const float* b_ih1  = (const float*)d_in[8];
  const float* b_hh1  = (const float*)d_in[9];
  const float* fc_w   = (const float*)d_in[10];
  const float* fc_b   = (const float*)d_in[11];
  float* out = (float*)d_out;

  // 3 x (384*128) weight matrices + 2 x (64*128) hidden states, all bf16:
  // (147456 + 16384) * 2 = 327680 bytes = exactly the 320 KB WGP LDS.
  const size_t lds_bytes = (size_t)(3 * 384 * 128 + 2 * 64 * 128) * 2;
  hipFuncSetAttribute((const void*)gru2_fused_kernel,
                      hipFuncAttributeMaxDynamicSharedMemorySize, (int)lds_bytes);

  gru2_fused_kernel<<<1, NTHREADS, lds_bytes, stream>>>(
      data, hidden, w_ih0, w_hh0, b_ih0, b_hh0,
      w_ih1, w_hh1, b_ih1, b_hh1, fc_w, fc_b, out);
}
